// ShadowKVAttention_33062658245085
// MI455X (gfx1250) — matrix operand order repaired and checked by reading, not compiled, not run on hardware
//
#include <hip/hip_runtime.h>
#include <math.h>

// ---------------------------------------------------------------------------
// ShadowKV decode attention for MI455X (gfx1250, wave32, WMMA + async-LDS).
// Bandwidth-bound (~250 MB/call @ 23.3 TB/s). WMMA for U_sel @ Vr; V-tile
// gathers go through GLOBAL_LOAD_ASYNC_TO_LDS_B128 (ASYNCcnt) so their
// latency overlaps the WMMA/RoPE/score phases of each 16-position tile.
// Workspace (floats): q[16384] kcur[4096] vcur[4096] attn[16384]
//                     partials[4*8*8*4*130 = 133120]  -> ~700 KB total.
// ---------------------------------------------------------------------------

typedef __attribute__((ext_vector_type(16))) _Float16 v16h;
typedef __attribute__((ext_vector_type(4)))  _Float16 v4h;
typedef __attribute__((ext_vector_type(8)))  float    v8f;
typedef __attribute__((ext_vector_type(4)))  float    v4f;

#define B_      4
#define DM      4096
#define H_      32
#define HKV_    8
#define HD_     128
#define GROUPS_ 4
#define CTX_    32768
#define RANK_   160
#define SSEL_   2048
#define NSPLIT  8
#define POSPS   (SSEL_ / NSPLIT)   // 256 positions per split
#define TILE_   16
#define NTILES  (POSPS / TILE_)    // 16 tiles per split
#define KCHUNKS (RANK_ / 32)       // 5 WMMA K-steps
#define SCALE_  0.088388347648318447f  // 128^-0.5

// Async copy of 16 bytes global -> LDS (ASYNCcnt path, gfx1250).
__device__ __forceinline__ void async_copy_b128(unsigned lds_off, const float* gaddr)
{
    asm volatile("global_load_async_to_lds_b128 %0, %1, off"
                 :: "v"(lds_off), "v"(gaddr) : "memory");
}
__device__ __forceinline__ void wait_async0()
{
#if __has_builtin(__builtin_amdgcn_s_wait_asynccnt)
    __builtin_amdgcn_s_wait_asynccnt(0);
#else
    asm volatile("s_wait_asynccnt 0x0" ::: "memory");
#endif
}

// --------------------------- K1: QKV projections ---------------------------
// One block per output column (q:4096, k:1024, v:1024 -> 6144 blocks).
// b128 weight loads; weight row reused across the 4 batch tokens.
__global__ __launch_bounds__(256) void qkv_gemv(
    const float* __restrict__ hs, const float* __restrict__ qw,
    const float* __restrict__ kw, const float* __restrict__ vw,
    float* __restrict__ q, float* __restrict__ kc, float* __restrict__ vc)
{
    int col = blockIdx.x;
    const float* wrow;
    float* out;
    int ocol, ostride;
    if (col < 4096)      { wrow = qw + (size_t)col * DM;          out = q;  ocol = col;        ostride = H_ * HD_; }
    else if (col < 5120) { wrow = kw + (size_t)(col - 4096) * DM; out = kc; ocol = col - 4096; ostride = HKV_ * HD_; }
    else                 { wrow = vw + (size_t)(col - 5120) * DM; out = vc; ocol = col - 5120; ostride = HKV_ * HD_; }

    float acc[B_] = {0.f, 0.f, 0.f, 0.f};
    for (int i = threadIdx.x * 4; i < DM; i += 1024) {
        v4f w = *(const v4f*)&wrow[i];
#pragma unroll
        for (int b = 0; b < B_; ++b) {
            v4f x = *(const v4f*)&hs[b * DM + i];
            acc[b] += w[0]*x[0] + w[1]*x[1] + w[2]*x[2] + w[3]*x[3];
        }
    }
    __shared__ float red[B_][256];
#pragma unroll
    for (int b = 0; b < B_; ++b) red[b][threadIdx.x] = acc[b];
    __syncthreads();
    for (int s = 128; s > 0; s >>= 1) {
        if (threadIdx.x < s)
#pragma unroll
            for (int b = 0; b < B_; ++b) red[b][threadIdx.x] += red[b][threadIdx.x + s];
        __syncthreads();
    }
    if (threadIdx.x < B_) out[threadIdx.x * ostride + ocol] = red[threadIdx.x][0];
}

// ------------------- K2: RoPE on current-token q and k ---------------------
__global__ __launch_bounds__(64) void rope_current(
    float* __restrict__ q, float* __restrict__ kc,
    const float* __restrict__ cosq, const float* __restrict__ sinq)
{
    int id = blockIdx.x;              // 0..B*H-1 = q heads, then B*HKV k heads
    float* v; int b;
    if (id < B_ * H_) { v = q + (size_t)id * HD_; b = id / H_; }
    else { int i2 = id - B_ * H_; v = kc + (size_t)i2 * HD_; b = i2 / HKV_; }
    int d  = threadIdx.x;             // 0..63 (pair d, d+64)
    float c0 = cosq[b * HD_ + d],      s0 = sinq[b * HD_ + d];
    float c1 = cosq[b * HD_ + d + 64], s1 = sinq[b * HD_ + d + 64];
    float x1 = v[d], x2 = v[d + 64];
    v[d]      = x1 * c0 - x2 * s0;
    v[d + 64] = x2 * c1 + x1 * s1;
}

// ------- K3: flash-decoding split: k_sel GEMM (WMMA) + RoPE + softmax ------
// grid = B*HKV*NSPLIT (256 blocks), 256 threads = 8 waves.
// Wave w computes the 16-column HD slice [16w,16w+16) of each 16-pos K tile.
__global__ __launch_bounds__(256) void flash_split(
    const float* __restrict__ U, const float* __restrict__ Vr,
    const float* __restrict__ vcache, const float* __restrict__ cosc,
    const float* __restrict__ sinc, const int* __restrict__ rpi,
    const float* __restrict__ qws, float* __restrict__ part)
{
    const int bid   = blockIdx.x;
    const int split = bid % NSPLIT;
    const int bh    = bid / NSPLIT;      // b*HKV + h
    const int b     = bh / HKV_;
    const int h     = bh % HKV_;

    const int tid      = threadIdx.x;
    const int lane     = tid & 31;
    const int wave     = tid >> 5;
    const int laneM    = lane & 15;
    const int laneHalf = lane >> 4;

    __shared__ __attribute__((aligned(16))) int      s_idx[TILE_];
    __shared__ __attribute__((aligned(16))) float    q_lds[GROUPS_][HD_];
    __shared__ __attribute__((aligned(16))) _Float16 u_lds[TILE_][RANK_ + 8]; // stride 168 halves (336 B)
    __shared__ __attribute__((aligned(16))) float    k_raw[TILE_][HD_ + 4];   // stride 132 (528 B)
    __shared__ __attribute__((aligned(16))) float    k_rope[TILE_][HD_ + 4];
    __shared__ __attribute__((aligned(16))) float    v_tile[TILE_][HD_ + 4];
    __shared__ __attribute__((aligned(16))) float    s_lds[GROUPS_][TILE_];

    // q (already RoPE'd) for the 4 query heads of this KV group (b128 loads)
    for (int e4 = tid; e4 < GROUPS_ * HD_ / 4; e4 += 256) {
        int g = e4 >> 5, d4 = (e4 & 31) * 4;
        *(v4f*)&q_lds[g][d4] =
            *(const v4f*)&qws[((size_t)b * H_ + h * GROUPS_ + g) * HD_ + d4];
    }

    // Preload B fragments: Vr 32x16 K-chunks (f32->f16), reused for all tiles.
    // B k order chosen here (same as A, not AMD's table order): lane 0-15 -> N=lane, K=j; lane 16-31 -> K=16+j.
    v16h bfrag[KCHUNKS];
    {
        const float* vr = Vr + (size_t)bh * RANK_ * HD_;
#pragma unroll
        for (int kc = 0; kc < KCHUNKS; ++kc) {
#pragma unroll
            for (int j = 0; j < 16; ++j) {
                int k   = kc * 32 + laneHalf * 16 + j;
                int col = wave * 16 + laneM;
                bfrag[kc][j] = (_Float16)vr[(size_t)k * HD_ + col];
            }
        }
    }

    float m_run = -1e30f, l_run = 0.f, acc0 = 0.f, acc1 = 0.f;
    const int hme = tid >> 6;                 // head owned by this thread
    const int d0  = (tid & 63) * 2, d1 = d0 + 1;

    const int    base_pos = split * POSPS;
    const int*   idxp = rpi    + (size_t)bh * SSEL_;
    const float* up   = U      + (size_t)bh * CTX_ * RANK_;
    const float* vp   = vcache + (size_t)bh * CTX_ * HD_;

    for (int t = 0; t < NTILES; ++t) {
        __syncthreads();                       // previous tile fully consumed
        if (tid < TILE_) s_idx[tid] = idxp[base_pos + t * TILE_ + tid];
        if (t + 1 < NTILES && tid == 32)
            __builtin_prefetch(idxp + base_pos + (t + 1) * TILE_, 0, 0);
        __syncthreads();

        // Kick off V-row gather straight into LDS (async, ASYNCcnt).
        // Consumed only in the accumulation phase -> full latency overlap.
        for (int e4 = tid; e4 < TILE_ * HD_ / 4; e4 += 256) {   // 512 x b128
            int r = e4 >> 5, d4 = (e4 & 31) * 4;
            async_copy_b128((unsigned)(uintptr_t)&v_tile[r][d4],
                            &vp[(size_t)s_idx[r] * HD_ + d4]);
        }

        // Gather U rows, f32 -> f16 into LDS (b128 global, b64 LDS stores).
        for (int e4 = tid; e4 < TILE_ * (RANK_ / 4); e4 += 256) { // 640 items
            int r = e4 / (RANK_ / 4), k4 = e4 - r * (RANK_ / 4);
            v4f u = *(const v4f*)&up[(size_t)s_idx[r] * RANK_ + k4 * 4];
            v4h uh;
            uh[0] = (_Float16)u[0]; uh[1] = (_Float16)u[1];
            uh[2] = (_Float16)u[2]; uh[3] = (_Float16)u[3];
            *(v4h*)&u_lds[r][k4 * 4] = uh;
        }
        __syncthreads();

        // K tile (16 pos x 128 dim) = U_sel(16xRANK) x Vr(RANKx128) via WMMA.
        // A k order chosen here (same as B, not AMD's table order): lane M=lane&15; half j -> K=16*laneHalf+j.
        v8f cacc = {};
#pragma unroll
        for (int kc = 0; kc < KCHUNKS; ++kc) {
            v16h afrag;
#pragma unroll
            for (int j = 0; j < 16; ++j) {
                int k = kc * 32 + (laneHalf << 4) + j;
                afrag[j] = u_lds[laneM][k];
            }
            cacc = __builtin_amdgcn_wmma_f32_16x16x32_f16(
                false, afrag, false, bfrag[kc], (short)0, cacc, false, false);
        }
        // C/D layout: VGPR r -> row r (lanes 0-15) / row r+8 (lanes 16-31).
#pragma unroll
        for (int r = 0; r < 8; ++r)
            k_raw[r + laneHalf * 8][wave * 16 + laneM] = cacc[r];
        __syncthreads();

        // RoPE on the K tile with gathered cos/sin rows (b128, 4-wide chunks).
        for (int e4 = tid; e4 < TILE_ * HD_ / 4; e4 += 256) {    // 512 items
            int r = e4 >> 5, dq = (e4 & 31) * 4;
            int p = s_idx[r];
            v4f c = *(const v4f*)&cosc[(size_t)p * HD_ + dq];
            v4f s = *(const v4f*)&sinc[(size_t)p * HD_ + dq];
            v4f x = *(const v4f*)&k_raw[r][dq];
            v4f o = *(const v4f*)&k_raw[r][dq ^ 64];
            float sgn = (dq < 64) ? -1.f : 1.f;
            v4f kr = x * c + (sgn * o) * s;
            *(v4f*)&k_rope[r][dq] = kr;
        }
        __syncthreads();

        // Scores: 64 (head,pos) dots of length 128, 4 lanes per dot.
        {
            int item = tid >> 2;               // 0..63
            int g = item >> 4, p = item & 15;
            int sub = tid & 3;
            float ps = 0.f;
#pragma unroll
            for (int i = 0; i < 32; ++i)
                ps += q_lds[g][sub * 32 + i] * k_rope[p][sub * 32 + i];
            ps += __shfl_xor(ps, 1, 32);
            ps += __shfl_xor(ps, 2, 32);
            if (sub == 0) s_lds[g][p] = ps * SCALE_;
        }
        wait_async0();                         // V tile landed in LDS
        __syncthreads();

        // Online softmax + PV accumulation (m,l redundant per 64-thread head).
        {
            float tmax = -1e30f;
#pragma unroll
            for (int p = 0; p < TILE_; ++p) tmax = fmaxf(tmax, s_lds[hme][p]);
            float m_new = fmaxf(m_run, tmax);
            float corr  = __expf(m_run - m_new);
            l_run *= corr; acc0 *= corr; acc1 *= corr;
#pragma unroll
            for (int p = 0; p < TILE_; ++p) {
                float pr = __expf(s_lds[hme][p] - m_new);
                l_run += pr;
                acc0  += pr * v_tile[p][d0];
                acc1  += pr * v_tile[p][d1];
            }
            m_run = m_new;
        }
    }

    // Partials: [bh][split][g] -> {m, l, acc[128]}
    float* pp = part + (((size_t)bh * NSPLIT + split) * GROUPS_ + hme) * (HD_ + 2);
    if ((tid & 63) == 0) { pp[0] = m_run; pp[1] = l_run; }
    pp[2 + d0] = acc0;
    pp[2 + d1] = acc1;
}

// ---------------- K4: combine splits + current token per head --------------
__global__ __launch_bounds__(128) void combine_partials(
    const float* __restrict__ part, const float* __restrict__ qws,
    const float* __restrict__ kcur, const float* __restrict__ vcur,
    float* __restrict__ attn)
{
    int bid = blockIdx.x;                   // b*H + qh
    int b = bid / H_, qh = bid % H_;
    int hkv = qh / GROUPS_, g = qh % GROUPS_;
    int d = threadIdx.x;                    // 0..127

    __shared__ float red[128];
    float qv = qws[(size_t)bid * HD_ + d];
    float kv = kcur[((size_t)b * HKV_ + hkv) * HD_ + d];
    red[d] = qv * kv;
    __syncthreads();
    for (int s = 64; s > 0; s >>= 1) {
        if (d < s) red[d] += red[d + s];
        __syncthreads();
    }
    float s_cur = red[0] * SCALE_;

    const float* pb = part + ((size_t)(b * HKV_ + hkv)) * NSPLIT * GROUPS_ * (HD_ + 2);
    float m = s_cur;
#pragma unroll
    for (int s = 0; s < NSPLIT; ++s)
        m = fmaxf(m, pb[(s * GROUPS_ + g) * (HD_ + 2)]);
    float wc = __expf(s_cur - m);
    float l  = wc;
    float o  = wc * vcur[((size_t)b * HKV_ + hkv) * HD_ + d];
#pragma unroll
    for (int s = 0; s < NSPLIT; ++s) {
        const float* ps = pb + (s * GROUPS_ + g) * (HD_ + 2);
        float w = __expf(ps[0] - m);
        l += w * ps[1];
        o += w * ps[2 + d];
    }
    attn[(size_t)bid * HD_ + d] = o / l;
}

// ------------------------- K5: output projection ---------------------------
__global__ __launch_bounds__(256) void o_gemv(
    const float* __restrict__ attn, const float* __restrict__ ow,
    float* __restrict__ out)
{
    int m = blockIdx.x;
    const float* wrow = ow + (size_t)m * DM;
    float acc[B_] = {0.f, 0.f, 0.f, 0.f};
    for (int i = threadIdx.x * 4; i < DM; i += 1024) {
        v4f w = *(const v4f*)&wrow[i];
#pragma unroll
        for (int b = 0; b < B_; ++b) {
            v4f x = *(const v4f*)&attn[b * DM + i];
            acc[b] += w[0]*x[0] + w[1]*x[1] + w[2]*x[2] + w[3]*x[3];
        }
    }
    __shared__ float red[B_][256];
#pragma unroll
    for (int b = 0; b < B_; ++b) red[b][threadIdx.x] = acc[b];
    __syncthreads();
    for (int s = 128; s > 0; s >>= 1) {
        if (threadIdx.x < s)
#pragma unroll
            for (int b = 0; b < B_; ++b) red[b][threadIdx.x] += red[b][threadIdx.x + s];
        __syncthreads();
    }
    if (threadIdx.x < B_) out[threadIdx.x * DM + m] = red[threadIdx.x][0];
}

// ---------------------------------------------------------------------------
extern "C" void kernel_launch(void* const* d_in, const int* in_sizes, int n_in,
                              void* d_out, int out_size, void* d_ws, size_t ws_size,
                              hipStream_t stream)
{
    const float* hs   = (const float*)d_in[0];
    const float* qw   = (const float*)d_in[1];
    const float* kw   = (const float*)d_in[2];
    const float* vw   = (const float*)d_in[3];
    const float* ow   = (const float*)d_in[4];
    const float* cosq = (const float*)d_in[5];
    const float* sinq = (const float*)d_in[6];
    const float* U    = (const float*)d_in[7];
    const float* Vr   = (const float*)d_in[8];
    const float* vch  = (const float*)d_in[9];
    const float* cosc = (const float*)d_in[10];
    const float* sinc = (const float*)d_in[11];
    const int*   rpi  = (const int*)  d_in[12];
    float* out = (float*)d_out;

    float* ws   = (float*)d_ws;
    float* q    = ws;                 // B*H*HD      = 16384
    float* kc   = ws + 16384;         // B*HKV*HD    = 4096
    float* vc   = ws + 20480;         // B*HKV*HD    = 4096
    float* attn = ws + 24576;         // B*H*HD      = 16384
    float* part = ws + 40960;         // B*HKV*NSPLIT*GROUPS*(HD+2) = 133120

    qkv_gemv<<<H_ * HD_ + 2 * HKV_ * HD_, 256, 0, stream>>>(hs, qw, kw, vw, q, kc, vc);
    rope_current<<<B_ * (H_ + HKV_), 64, 0, stream>>>(q, kc, cosq, sinq);
    flash_split<<<B_ * HKV_ * NSPLIT, 256, 0, stream>>>(U, Vr, vch, cosc, sinc, rpi, q, part);
    combine_partials<<<B_ * H_, 128, 0, stream>>>(part, q, kc, vc, attn);
    o_gemv<<<DM, 256, 0, stream>>>(attn, ow, out);
}
